// GCN_44263932952820
// MI455X (gfx1250) — matrix operand order repaired and checked by reading, not compiled, not run on hardware
//
#include <hip/hip_runtime.h>
#include <hip/hip_bf16.h>

typedef __attribute__((ext_vector_type(16))) __bf16 v16bf;
typedef __attribute__((ext_vector_type(8)))  float  v8f;

#define GN   100000
#define GE   1600000
#define GFIN 128
#define GHID 48
#define GNC  40
#define GEPS 1e-5f
#define GSLOPE 0.01f

// ---------------------------------------------------------------------------
// Pre-pack weights (fp32 -> bf16) directly into per-lane WMMA B-fragment
// order: P[chunk][tile][lane][16], so the GEMM loads one contiguous 32B
// v16bf per fragment.  B layout (16-bit, 32x16): lane half 0 holds K {0..7,16..23},
// half 1 holds K {8..15,24..31}, 2 values per VGPR; lane idx = output column.
// ---------------------------------------------------------------------------
__device__ inline void GCN_pack_one(const float* __restrict__ W, int ldW,
                                    int Kreal, int Cvalid, int nchunks,
                                    __bf16* __restrict__ P, int tid)
{
    int total = nchunks * 3 * 32 * 16;
    for (int i = tid; i < total; i += 256) {
        int e    = i & 15;
        int lane = (i >> 4) & 31;
        int t    = (i >> 9) % 3;
        int ch   = i / (512 * 3);
        int half = lane >> 4, idx = lane & 15;
        int p = e >> 1, lo = e & 1;
        int k = ch * 32 + (e >> 3) * 16 + half * 8 + (e & 7);
        int n = t * 16 + idx;
        float v = (k < Kreal && n < Cvalid) ? W[k * ldW + n] : 0.f;
        P[i] = (__bf16)v;
    }
}

__global__ __launch_bounds__(256) void GCN_pack_weights(
    const float* __restrict__ Wf, const float* __restrict__ Wc1,
    const float* __restrict__ Wc2, const float* __restrict__ Wout,
    const float* __restrict__ bout,
    __bf16* WfP, __bf16* Wc1P, __bf16* Wc2P, __bf16* WoutP, float* bout_p)
{
    int tid = threadIdx.x;
    GCN_pack_one(Wf,   48, 128, 48, 4, WfP,   tid);   // 128x48
    GCN_pack_one(Wc1,  48,  48, 48, 2, Wc1P,  tid);   // 48x48 -> Kpad 64
    GCN_pack_one(Wc2,  48,  48, 48, 2, Wc2P,  tid);   // 48x48 -> Kpad 64
    GCN_pack_one(Wout, 40, 144, 40, 5, WoutP, tid);   // 144x40 -> 160x48
    if (tid < 48) bout_p[tid] = (tid < 40) ? bout[tid] : 0.f;
}

// ---------------------------------------------------------------------------
// Degree / normalization
// ---------------------------------------------------------------------------
__global__ __launch_bounds__(256) void GCN_deg_init(float* deg) {
    int i = blockIdx.x * 256 + threadIdx.x;
    if (i < GN) deg[i] = 1.0f;               // self-loop weight 1
}
__global__ __launch_bounds__(256) void GCN_deg_acc(
    const long long* __restrict__ dst, const float* __restrict__ ew, float* deg) {
    int e = blockIdx.x * 256 + threadIdx.x;
    if (e < GE) atomicAdd(&deg[dst[e]], ew[e]);
}
__global__ __launch_bounds__(256) void GCN_dinv(const float* deg, float* dinv) {
    int i = blockIdx.x * 256 + threadIdx.x;
    if (i < GN) dinv[i] = (deg[i] > 0.f) ? __frsqrt_rn(deg[i]) : 0.f;
}

// ---------------------------------------------------------------------------
// A-fragment loader (16-bit A 16x32, ISA 7.12.2): lane half 0 holds
// K {0..7,16..23}, half 1 holds K {8..15,24..31}, 2 per VGPR.
// KREAL/chunk compile-time => guard folds to a constant (no divergence);
// valid for KREAL multiple of 16 (128, 48, 144 all qualify).
// ---------------------------------------------------------------------------
template <int KREAL>
__device__ inline v16bf GCN_load_afrag(const float* __restrict__ Arow,
                                       int kc, int half)
{
    v16bf a;
#pragma unroll
    for (int p = 0; p < 8; ++p) {
        const int base = (p < 4) ? 0 : 16;
        float2 f2;
        if (kc + base + 15 < KREAL)
            f2 = *(const float2*)(Arow + kc + base + half * 8 + (p & 3) * 2);
        else
            f2 = make_float2(0.f, 0.f);
        a[2 * p]     = (__bf16)f2.x;
        a[2 * p + 1] = (__bf16)f2.y;
    }
    return a;
}

// ---------------------------------------------------------------------------
// WMMA bf16 GEMM: Out[M x 48] = A[M x KREAL] * Bpacked  (+bias)(+lrelu)
// 128 threads = 4 waves; each wave owns a 16-row tile x 48 cols (3 accs).
// ---------------------------------------------------------------------------
template <int KREAL, int KPAD>
__global__ __launch_bounds__(128) void GCN_wmma_gemm(
    const float* __restrict__ A, int ldA, int colA0,
    const __bf16* __restrict__ BP,
    const float* __restrict__ bias, int do_bias, int do_lrelu,
    float* __restrict__ Out, int ldOut, int colOut0)
{
    int lane = threadIdx.x & 31;
    int wave = threadIdx.x >> 5;
    size_t row0 = ((size_t)blockIdx.x * 4 + wave) * 16;
    if (row0 >= GN) return;                      // wave-uniform guard
    int half = lane >> 4, idx = lane & 15;

    const v16bf* Bv = (const v16bf*)BP;
    v8f acc[3] = { v8f{}, v8f{}, v8f{} };
    const float* Arow = A + (row0 + idx) * (size_t)ldA + colA0;

#pragma unroll
    for (int ch = 0; ch < KPAD / 32; ++ch) {
        v16bf a = GCN_load_afrag<KREAL>(Arow, ch * 32, half);
#pragma unroll
        for (int t = 0; t < 3; ++t) {
            v16bf b = Bv[(ch * 3 + t) * 32 + lane];
            acc[t] = __builtin_amdgcn_wmma_f32_16x16x32_bf16(
                false, a, false, b, (short)0, acc[t], false, false);
        }
    }

#pragma unroll
    for (int t = 0; t < 3; ++t) {
        int n = t * 16 + idx;
        float bv = do_bias ? bias[n] : 0.f;
#pragma unroll
        for (int r = 0; r < 8; ++r) {
            size_t m = row0 + r + half * 8;
            float v = acc[t][r] + bv;
            if (do_lrelu) v = (v >= 0.f) ? v : GSLOPE * v;
            Out[m * (size_t)ldOut + colOut0 + n] = v;
        }
    }
}

// ---------------------------------------------------------------------------
// Output GEMM: concat(h1,h2,h3)[N x 144] @ Wout(packed 160x48) + b, then
// row-wise log_softmax over the 40 valid classes -> d_out [N x 40].
// C/D layout: lane idx = column, vgpr r = row r + half*8; each output row
// lives in one 16-lane half -> xor-mask shuffle reduction.
// ---------------------------------------------------------------------------
__global__ __launch_bounds__(128) void GCN_wmma_out(
    const float* __restrict__ H, const __bf16* __restrict__ BP,
    const float* __restrict__ bias, float* __restrict__ out)
{
    int lane = threadIdx.x & 31;
    int wave = threadIdx.x >> 5;
    size_t row0 = ((size_t)blockIdx.x * 4 + wave) * 16;
    if (row0 >= GN) return;
    int half = lane >> 4, idx = lane & 15;

    const v16bf* Bv = (const v16bf*)BP;
    v8f acc[3] = { v8f{}, v8f{}, v8f{} };
    const float* Arow = H + (row0 + idx) * (size_t)144;

#pragma unroll
    for (int ch = 0; ch < 5; ++ch) {
        v16bf a = GCN_load_afrag<144>(Arow, ch * 32, half);
#pragma unroll
        for (int t = 0; t < 3; ++t) {
            v16bf b = Bv[(ch * 3 + t) * 32 + lane];
            acc[t] = __builtin_amdgcn_wmma_f32_16x16x32_bf16(
                false, a, false, b, (short)0, acc[t], false, false);
        }
    }

    bool v2 = (32 + idx) < 40;
#pragma unroll
    for (int r = 0; r < 8; ++r) {
        float c0 = acc[0][r] + bias[idx];
        float c1 = acc[1][r] + bias[16 + idx];
        float c2 = acc[2][r] + bias[32 + idx];
        float mx = fmaxf(c0, c1);
        if (v2) mx = fmaxf(mx, c2);
#pragma unroll
        for (int s = 8; s >= 1; s >>= 1) mx = fmaxf(mx, __shfl_xor(mx, s, 32));
        float se = __expf(c0 - mx) + __expf(c1 - mx) + (v2 ? __expf(c2 - mx) : 0.f);
#pragma unroll
        for (int s = 8; s >= 1; s >>= 1) se += __shfl_xor(se, s, 32);
        float lse = __logf(se) + mx;

        size_t m = row0 + r + half * 8;
        out[m * 40 + idx]      = c0 - lse;
        out[m * 40 + 16 + idx] = c1 - lse;
        if (v2) out[m * 40 + 32 + idx] = c2 - lse;
    }
}

// ---------------------------------------------------------------------------
// Initialize a 48-wide column slice of H to the conv bias (accumulator base).
// ---------------------------------------------------------------------------
__global__ __launch_bounds__(256) void GCN_init_slice(
    float* __restrict__ H, int ldH, int col0, const float* __restrict__ b)
{
    long long i = (long long)blockIdx.x * 256 + threadIdx.x;
    if (i >= (long long)GN * 48) return;
    int row = (int)(i / 48), c = (int)(i % 48);
    H[(size_t)row * ldH + col0 + c] = b[c];
}

// ---------------------------------------------------------------------------
// Edge scatter: H[dst, slice] += T[src] * (dinv[src]*w*dinv[dst]).
// Edges e < GE from edge_index; e >= GE are the self-loops (w=1).
// ---------------------------------------------------------------------------
__global__ __launch_bounds__(256) void GCN_scatter(
    const float* __restrict__ T, const long long* __restrict__ src,
    const long long* __restrict__ dst, const float* __restrict__ ew,
    const float* __restrict__ dinv, float* __restrict__ H, int ldH, int col0)
{
    long long e = (long long)blockIdx.x * 256 + threadIdx.x;
    if (e >= (long long)GE + GN) return;
    long long s, d; float w;
    if (e < GE) { s = src[e]; d = dst[e]; w = ew[e]; }
    else        { s = d = e - GE;         w = 1.0f;  }
    float norm = dinv[s] * w * dinv[d];
    const float4* tr = (const float4*)(T + (size_t)s * 48);
    float* outp = H + (size_t)d * ldH + col0;
#pragma unroll
    for (int q = 0; q < 12; ++q) {
        float4 v = tr[q];
        atomicAdd(outp + 4 * q + 0, v.x * norm);
        atomicAdd(outp + 4 * q + 1, v.y * norm);
        atomicAdd(outp + 4 * q + 2, v.z * norm);
        atomicAdd(outp + 4 * q + 3, v.w * norm);
    }
}

// ---------------------------------------------------------------------------
// Fused leaky-relu (optional, in place) + per-column sum / sumsq for BN.
// stats[0:48] = sum, stats[48:96] = sumsq. LDS reduce -> global atomics.
// ---------------------------------------------------------------------------
__global__ __launch_bounds__(96) void GCN_zero_stats(float* stats) {
    stats[threadIdx.x] = 0.f;
}
__global__ __launch_bounds__(256) void GCN_lrelu_stats(
    float* __restrict__ H, int ldH, int col0, int do_lrelu, float* __restrict__ stats)
{
    __shared__ float ls[96];
    int tid = threadIdx.x;
    if (tid < 96) ls[tid] = 0.f;
    __syncthreads();
    long long i = (long long)blockIdx.x * 256 + tid;
    if (i < (long long)GN * 48) {
        int row = (int)(i / 48), c = (int)(i % 48);
        float* p = H + (size_t)row * ldH + col0 + c;
        float v = *p;
        if (do_lrelu) { v = (v >= 0.f) ? v : GSLOPE * v; *p = v; }
        atomicAdd(&ls[c], v);
        atomicAdd(&ls[48 + c], v * v);
    }
    __syncthreads();
    if (tid < 96) atomicAdd(&stats[tid], ls[tid]);
}
__global__ __launch_bounds__(256) void GCN_bn_apply(
    float* __restrict__ H, int ldH, int col0, const float* __restrict__ stats,
    const float* __restrict__ g, const float* __restrict__ b)
{
    long long i = (long long)blockIdx.x * 256 + threadIdx.x;
    if (i >= (long long)GN * 48) return;
    int row = (int)(i / 48), c = (int)(i % 48);
    float mu  = stats[c] * (1.0f / GN);
    float var = stats[48 + c] * (1.0f / GN) - mu * mu;
    float* p = H + (size_t)row * ldH + col0 + c;
    *p = g[c] * (*p - mu) * __frsqrt_rn(var + GEPS) + b[c];
}

// ---------------------------------------------------------------------------
extern "C" void kernel_launch(void* const* d_in, const int* in_sizes, int n_in,
                              void* d_out, int out_size, void* d_ws, size_t ws_size,
                              hipStream_t stream) {
    const float*     x      = (const float*)d_in[0];
    const long long* ei     = (const long long*)d_in[1];   // int64 [2,E]
    const float*     ew     = (const float*)d_in[2];
    const float*     Wf     = (const float*)d_in[3];
    const float*     bf     = (const float*)d_in[4];
    const float*     bn1g   = (const float*)d_in[5];
    const float*     bn1b   = (const float*)d_in[6];
    const float*     Wc1    = (const float*)d_in[7];
    const float*     bc1    = (const float*)d_in[8];
    const float*     bn2g   = (const float*)d_in[9];
    const float*     bn2b   = (const float*)d_in[10];
    const float*     Wc2    = (const float*)d_in[11];
    const float*     bc2    = (const float*)d_in[12];
    const float*     bn3g   = (const float*)d_in[13];
    const float*     bn3b   = (const float*)d_in[14];
    const float*     Wout   = (const float*)d_in[15];
    const float*     bout   = (const float*)d_in[16];
    float* out = (float*)d_out;

    const long long* esrc = ei;
    const long long* edst = ei + GE;

    char* ws = (char*)d_ws;
    size_t off = 0;
    auto take = [&](size_t bytes) { size_t o = off; off = (off + bytes + 255) & ~(size_t)255; return o; };
    float*  H      = (float*)(ws + take((size_t)GN * 144 * 4));
    float*  T      = (float*)(ws + take((size_t)GN * 48 * 4));
    float*  deg    = (float*)(ws + take((size_t)GN * 4));
    float*  dinv   = (float*)(ws + take((size_t)GN * 4));
    float*  stats  = (float*)(ws + take(96 * 4));
    __bf16* WfP    = (__bf16*)(ws + take(4 * 3 * 512 * 2));
    __bf16* Wc1P   = (__bf16*)(ws + take(2 * 3 * 512 * 2));
    __bf16* Wc2P   = (__bf16*)(ws + take(2 * 3 * 512 * 2));
    __bf16* WoutP  = (__bf16*)(ws + take(5 * 3 * 512 * 2));
    float*  bout_p = (float*)(ws + take(48 * 4));
    (void)ws_size; (void)n_in; (void)in_sizes; (void)out_size;

    const int GB   = (GN / 16 + 3) / 4;             // 4 row-tiles per block
    const int GNC_ = (GN + 255) / 256;
    const int GE_  = (GE + 255) / 256;
    const int GEL  = (int)(((long long)GE + GN + 255) / 256);
    const int GELM = (int)(((long long)GN * 48 + 255) / 256);

    GCN_pack_weights<<<1, 256, 0, stream>>>(Wf, Wc1, Wc2, Wout, bout,
                                            WfP, Wc1P, Wc2P, WoutP, bout_p);
    // graph normalization (shared by both conv layers)
    GCN_deg_init<<<GNC_, 256, 0, stream>>>(deg);
    GCN_deg_acc<<<GE_, 256, 0, stream>>>(edst, ew, deg);
    GCN_dinv<<<GNC_, 256, 0, stream>>>(deg, dinv);

    // ---- layer 1: h1 = BN(lrelu(x @ Wf + bf)) -> H[:,0:48]
    GCN_wmma_gemm<128, 128><<<GB, 128, 0, stream>>>(x, GFIN, 0, WfP,
                                                    bf, 1, 1, H, 144, 0);
    GCN_zero_stats<<<1, 96, 0, stream>>>(stats);
    GCN_lrelu_stats<<<GELM, 256, 0, stream>>>(H, 144, 0, 0, stats);
    GCN_bn_apply<<<GELM, 256, 0, stream>>>(H, 144, 0, stats, bn1g, bn1b);

    // ---- layer 2: h2 = BN(lrelu(gcn_conv(h1))) -> H[:,48:96]
    GCN_wmma_gemm<48, 64><<<GB, 128, 0, stream>>>(H, 144, 0, Wc1P,
                                                  nullptr, 0, 0, T, 48, 0);
    GCN_init_slice<<<GELM, 256, 0, stream>>>(H, 144, 48, bc1);
    GCN_scatter<<<GEL, 256, 0, stream>>>(T, esrc, edst, ew, dinv, H, 144, 48);
    GCN_zero_stats<<<1, 96, 0, stream>>>(stats);
    GCN_lrelu_stats<<<GELM, 256, 0, stream>>>(H, 144, 48, 1, stats);
    GCN_bn_apply<<<GELM, 256, 0, stream>>>(H, 144, 48, stats, bn2g, bn2b);

    // ---- layer 3: h3 -> H[:,96:144]
    GCN_wmma_gemm<48, 64><<<GB, 128, 0, stream>>>(H, 144, 48, Wc2P,
                                                  nullptr, 0, 0, T, 48, 0);
    GCN_init_slice<<<GELM, 256, 0, stream>>>(H, 144, 96, bc2);
    GCN_scatter<<<GEL, 256, 0, stream>>>(T, esrc, edst, ew, dinv, H, 144, 96);
    GCN_zero_stats<<<1, 96, 0, stream>>>(stats);
    GCN_lrelu_stats<<<GELM, 256, 0, stream>>>(H, 144, 96, 1, stats);
    GCN_bn_apply<<<GELM, 256, 0, stream>>>(H, 144, 96, stats, bn3g, bn3b);

    // ---- output: log_softmax(concat @ Wout + bout)
    GCN_wmma_out<<<GB, 128, 0, stream>>>(H, WoutP, bout_p, out);
}
